// MultiHeadAttention_5145370821217
// MI455X (gfx1250) — matrix operand order repaired and checked by reading, not compiled, not run on hardware
//
#include <hip/hip_runtime.h>

typedef unsigned short u16;
typedef __attribute__((ext_vector_type(16))) __bf16 v16bf;
typedef __attribute__((ext_vector_type(8)))  float  v8f;

#define D_MODEL 1024
#define SEQ     2048
#define BATCH   2
#define NH      16
#define DH      64

// ---------- helpers ----------

__device__ __forceinline__ u16 f2bf(float f) {
  unsigned u = __float_as_uint(f);
  u += 0x7FFFu + ((u >> 16) & 1u);          // round-to-nearest-even
  return (u16)(u >> 16);
}
__device__ __forceinline__ unsigned pk2(float a, float b) {
  return (unsigned)f2bf(a) | ((unsigned)f2bf(b) << 16);
}

union FragU { uint4 q[2]; v16bf v; };

// 16 bf16 from two 8-element (16B) chunks
__device__ __forceinline__ v16bf ldfrag2(const u16* p0, const u16* p1) {
  FragU u;
  u.q[0] = *(const uint4*)p0;
  u.q[1] = *(const uint4*)p1;
  return u.v;
}
// 16 contiguous bf16 (32B)
__device__ __forceinline__ v16bf ldfrag(const u16* p) { return ldfrag2(p, p + 8); }

// exchange with the other 16-lane half of the wave
__device__ __forceinline__ float xhalf(float v, int lane) {
  return __int_as_float(
      __builtin_amdgcn_ds_bpermute(((lane ^ 16) << 2), __float_as_int(v)));
}

#define WMMA_BF16(a, b, c) \
  __builtin_amdgcn_wmma_f32_16x16x32_bf16(false, (a), false, (b), (short)0, (c), false, false)

// ---------- kernel 1: QKV projection  P = (X @ W^T + b) * oscale  (bf16, head-major) ----------
// grid: (32*16, 3), block: 256 (8 waves). Block tile 128(M) x 64(N), K chunk 32.
// For the Q projection, oscale = 1/sqrt(Dh) * log2(e): logits leave the score
// WMMAs already in the exp2 domain (saves per-iteration scaling in attention).
__global__ void __launch_bounds__(256) proj_kernel(
    const float* __restrict__ qi, const float* __restrict__ ki, const float* __restrict__ vi,
    const float* __restrict__ Wq, const float* __restrict__ bq,
    const float* __restrict__ Wk, const float* __restrict__ bk,
    const float* __restrict__ Wv, const float* __restrict__ bv,
    u16* __restrict__ Qh, u16* __restrict__ Kh, u16* __restrict__ Vh) {
  __shared__ u16 At[128 * 32];   // A tile, row-major [m][k], 8KB
  __shared__ u16 Bt[32 * 64];    // W^T tile, [n][k], 4KB

  const float* X;  const float* W;  const float* bias;  u16* dst;  float oscale;
  if (blockIdx.y == 0)      { X = qi; W = Wq; bias = bq; dst = Qh;
                              oscale = 0.125f * 1.44269504088896340736f; }
  else if (blockIdx.y == 1) { X = ki; W = Wk; bias = bk; dst = Kh; oscale = 1.0f; }
  else                      { X = vi; W = Wv; bias = bv; dst = Vh; oscale = 1.0f; }

  const int mBase = (int)(blockIdx.x >> 4) * 128;
  const int nBase = (int)(blockIdx.x & 15) * 64;
  const int tid  = threadIdx.x;
  const int lane = tid & 31;
  const int wid  = tid >> 5;
  const int wm   = (wid & 3) * 32;
  const int wn   = (wid >> 2) * 32;
  const int lr   = lane & 15;
  const int kbl  = (lane >> 4) * 8;

  const int arow = tid >> 1;
  const int ac0  = (tid & 1) * 16;
  const int bn   = tid >> 2;
  const int bk0  = (tid & 3) * 8;

  v8f acc[2][2] = {};

  for (int kb = 0; kb < D_MODEL; kb += 32) {
    {  // stage A: 128x32 f32 -> bf16, row-major
      const float* src = X + (size_t)(mBase + arow) * D_MODEL + kb + ac0;
      float4 f0 = ((const float4*)src)[0];
      float4 f1 = ((const float4*)src)[1];
      float4 f2 = ((const float4*)src)[2];
      float4 f3 = ((const float4*)src)[3];
      uint4 p0 = { pk2(f0.x, f0.y), pk2(f0.z, f0.w), pk2(f1.x, f1.y), pk2(f1.z, f1.w) };
      uint4 p1 = { pk2(f2.x, f2.y), pk2(f2.z, f2.w), pk2(f3.x, f3.y), pk2(f3.z, f3.w) };
      *(uint4*)&At[arow * 32 + ac0]     = p0;
      *(uint4*)&At[arow * 32 + ac0 + 8] = p1;
    }
    {  // stage B n-major: Bt[n][k] = W[nBase+n][kb+k]
      const float* src = W + (size_t)(nBase + bn) * D_MODEL + kb + bk0;
      float4 f0 = ((const float4*)src)[0];
      float4 f1 = ((const float4*)src)[1];
      u16 e[8] = { f2bf(f0.x), f2bf(f0.y), f2bf(f0.z), f2bf(f0.w),
                   f2bf(f1.x), f2bf(f1.y), f2bf(f1.z), f2bf(f1.w) };
#pragma unroll
      for (int j = 0; j < 8; ++j) Bt[bn * 32 + bk0 + j] = e[j];
    }
    __syncthreads();

    v16bf af0 = ldfrag2(&At[(wm + lr) * 32 + kbl],      &At[(wm + lr) * 32 + kbl + 16]);
    v16bf af1 = ldfrag2(&At[(wm + 16 + lr) * 32 + kbl], &At[(wm + 16 + lr) * 32 + kbl + 16]);
    v16bf bf0 = ldfrag2(&Bt[(wn + lr) * 32 + kbl],      &Bt[(wn + lr) * 32 + kbl + 16]);
    v16bf bf1 = ldfrag2(&Bt[(wn + 16 + lr) * 32 + kbl], &Bt[(wn + 16 + lr) * 32 + kbl + 16]);

    acc[0][0] = WMMA_BF16(af0, bf0, acc[0][0]);
    acc[0][1] = WMMA_BF16(af0, bf1, acc[0][1]);
    acc[1][0] = WMMA_BF16(af1, bf0, acc[1][0]);
    acc[1][1] = WMMA_BF16(af1, bf1, acc[1][1]);
    __syncthreads();
  }

#pragma unroll
  for (int tm = 0; tm < 2; ++tm) {
#pragma unroll
    for (int tn = 0; tn < 2; ++tn) {
      const int nG = nBase + wn + tn * 16 + lr;
      const float bb = bias[nG];
      const int h = nG >> 6, d = nG & 63;
#pragma unroll
      for (int r = 0; r < 8; ++r) {
        const int mG = mBase + wm + tm * 16 + r + kbl;
        const int b = mG >> 11, s = mG & 2047;
        const size_t idx = ((((size_t)b * NH + h) * SEQ) + s) * DH + d;
        dst[idx] = f2bf((acc[tm][tn][r] + bb) * oscale);
      }
    }
  }
}

// ---------- kernel 2: flash attention, transposed-score formulation ----------
// grid: B*H*(S/128) = 512 blocks, 256 threads (8 waves); wave owns 16 queries.
// S^T = K (16k x 32d, A-frag straight from global) x Q^T (B-frag, staged once).
// Queries live in lanes -> softmax stats are per-lane scalars; only one xor-16
// bpermute per statistic. O^T += V^T x P^T; epilogue stores are float4.
__global__ void __launch_bounds__(256) attn_kernel(
    const u16* __restrict__ Qh, const u16* __restrict__ Kh, const u16* __restrict__ Vh,
    float* __restrict__ out) {
  __shared__ u16 Vt[DH * 32];        // V^T tile [d][key], 4KB
  __shared__ u16 Pb[8][1024];        // per-wave: Q^T (64x16) at init, then P^T (32x16)

  const int bh   = blockIdx.x >> 4;
  const int qBlk = blockIdx.x & 15;
  const int tid  = threadIdx.x;
  const int lane = tid & 31;
  const int wid  = tid >> 5;
  const int lr   = lane & 15;
  const int kbl  = (lane >> 4) * 8;     // 0 or 8: fragment half / row offset
  const int qTile = qBlk * 128 + wid * 16;

  const size_t base = (size_t)bh * SEQ * DH;
  const u16* Qp = Qh + base;
  const u16* Kp = Kh + base;
  const u16* Vp = Vh + base;

  // ---- stage Q^T once (wave-private region of Pb) ----
  u16* qt = Pb[wid];
  {
    const int dc = (lane >> 4) * 32;                    // d chunk 0..31 / 32..63
    const u16* src = Qp + (size_t)(qTile + lr) * DH + dc;
    uint4 a[4];
    a[0] = ((const uint4*)src)[0];
    a[1] = ((const uint4*)src)[1];
    a[2] = ((const uint4*)src)[2];
    a[3] = ((const uint4*)src)[3];
    const u16* e = (const u16*)a;
#pragma unroll
    for (int j = 0; j < 32; ++j) qt[lr * 64 + dc + j] = e[j];
  }
  const v16bf qb0 = ldfrag2(&qt[lr * 64 + kbl],      &qt[lr * 64 + kbl + 16]);   // d 0..31
  const v16bf qb1 = ldfrag2(&qt[lr * 64 + 32 + kbl], &qt[lr * 64 + 48 + kbl]);  // d 32..63

  v8f o[4] = {};
  float mrun = -1e30f, lsum = 0.f;

  const int ks  = tid >> 3;        // 0..31 local key for Vt staging
  const int kd0 = (tid & 7) * 8;   // d chunk for Vt staging

  for (int kt = 0; kt < SEQ; kt += 32) {
    {  // cooperative transpose of 32x64 V tile into Vt[d][key]
      const uint4 vv = *(const uint4*)(Vp + (size_t)(kt + ks) * DH + kd0);
      const u16* e = (const u16*)&vv;
#pragma unroll
      for (int j = 0; j < 8; ++j) Vt[(kd0 + j) * 32 + ks] = e[j];
    }
    __syncthreads();

    // prefetch next tile (clamped so we never touch past the workspace)
    const int ktn = (kt + 32 < SEQ) ? (kt + 32) : kt;
    __builtin_prefetch(Kp + (size_t)(ktn + lr) * DH, 0, 1);
    __builtin_prefetch(Vp + (size_t)(ktn + ks) * DH + kd0, 0, 1);

    // K A-fragments straight from global (natural [s][d] layout)
    const u16* krow0 = Kp + (size_t)(kt + lr) * DH;
    const u16* krow1 = Kp + (size_t)(kt + 16 + lr) * DH;
    const v16bf ka00 = ldfrag2(krow0 + kbl,      krow0 + kbl + 16);
    const v16bf ka01 = ldfrag2(krow0 + 32 + kbl, krow0 + 48 + kbl);
    const v16bf ka10 = ldfrag2(krow1 + kbl,      krow1 + kbl + 16);
    const v16bf ka11 = ldfrag2(krow1 + 32 + kbl, krow1 + 48 + kbl);

    // S^T tiles: s0 = keys 0..15, s1 = keys 16..31 (M=key, N=query)
    v8f s0 = {}, s1 = {};
    s0 = WMMA_BF16(ka00, qb0, s0);
    s0 = WMMA_BF16(ka01, qb1, s0);
    s1 = WMMA_BF16(ka10, qb0, s1);
    s1 = WMMA_BF16(ka11, qb1, s1);

    // per-query (= per-lane) online softmax; logits already log2-scaled
    float t = fmaxf(s0[0], s1[0]);
#pragma unroll
    for (int r = 1; r < 8; ++r) t = fmaxf(t, fmaxf(s0[r], s1[r]));
    t = fmaxf(t, xhalf(t, lane));
    const float mn = fmaxf(mrun, t);
    const float al = exp2f(mrun - mn);
    mrun = mn;

    float rs = 0.f;
#pragma unroll
    for (int r = 0; r < 8; ++r) {
      s0[r] = exp2f(s0[r] - mn);
      s1[r] = exp2f(s1[r] - mn);
      rs += s0[r] + s1[r];
    }
    rs += xhalf(rs, lane);
    lsum = lsum * al + rs;
#pragma unroll
    for (int dt = 0; dt < 4; ++dt)
#pragma unroll
      for (int r = 0; r < 8; ++r) o[dt][r] *= al;

    // P^T (C-layout f32) -> LDS bf16 [q][k], read back as B-fragment
    u16* pb = Pb[wid];
#pragma unroll
    for (int r = 0; r < 8; ++r) {
      pb[lr * 32 + r + kbl]        = f2bf(s0[r]);
      pb[lr * 32 + 16 + r + kbl]   = f2bf(s1[r]);
    }
    const v16bf pfrag = ldfrag2(&pb[lr * 32 + kbl], &pb[lr * 32 + kbl + 16]);

    // O^T += V^T x P^T  (A-frags from transposed Vt)
    const v16bf va0 = ldfrag2(&Vt[(lr)      * 32 + kbl], &Vt[(lr)      * 32 + kbl + 16]);
    const v16bf va1 = ldfrag2(&Vt[(16 + lr) * 32 + kbl], &Vt[(16 + lr) * 32 + kbl + 16]);
    const v16bf va2 = ldfrag2(&Vt[(32 + lr) * 32 + kbl], &Vt[(32 + lr) * 32 + kbl + 16]);
    const v16bf va3 = ldfrag2(&Vt[(48 + lr) * 32 + kbl], &Vt[(48 + lr) * 32 + kbl + 16]);
    o[0] = WMMA_BF16(va0, pfrag, o[0]);
    o[1] = WMMA_BF16(va1, pfrag, o[1]);
    o[2] = WMMA_BF16(va2, pfrag, o[2]);
    o[3] = WMMA_BF16(va3, pfrag, o[3]);
    __syncthreads();
  }

  // normalize and store f32 [B, S, D] with vector stores (O^T: d = dt*16 + r + kbl)
  const float inv = 1.0f / lsum;
  const int b = bh >> 4;
  const int head = bh & 15;
  float* orow = out + ((size_t)b * SEQ + qTile + lr) * D_MODEL + head * DH + kbl;
#pragma unroll
  for (int dt = 0; dt < 4; ++dt) {
    float4 lo = { o[dt][0] * inv, o[dt][1] * inv, o[dt][2] * inv, o[dt][3] * inv };
    float4 hi = { o[dt][4] * inv, o[dt][5] * inv, o[dt][6] * inv, o[dt][7] * inv };
    *(float4*)(orow + dt * 16)     = lo;
    *(float4*)(orow + dt * 16 + 4) = hi;
  }
}

// ---------- host launcher ----------
extern "C" void kernel_launch(void* const* d_in, const int* in_sizes, int n_in,
                              void* d_out, int out_size, void* d_ws, size_t ws_size,
                              hipStream_t stream) {
  (void)in_sizes; (void)n_in; (void)out_size; (void)ws_size;
  const float* q  = (const float*)d_in[0];
  const float* k  = (const float*)d_in[1];
  const float* v  = (const float*)d_in[2];
  const float* Wq = (const float*)d_in[3];
  const float* bq = (const float*)d_in[4];
  const float* Wk = (const float*)d_in[5];
  const float* bk = (const float*)d_in[6];
  const float* Wv = (const float*)d_in[7];
  const float* bv = (const float*)d_in[8];

  const size_t elems = (size_t)BATCH * SEQ * D_MODEL;  // 4M
  u16* Qh = (u16*)d_ws;
  u16* Kh = Qh + elems;
  u16* Vh = Kh + elems;

  dim3 pgrid((BATCH * SEQ / 128) * (D_MODEL / 64), 3, 1);  // (512, 3)
  proj_kernel<<<pgrid, 256, 0, stream>>>(q, k, v, Wq, bq, Wk, bk, Wv, bv, Qh, Kh, Vh);

  dim3 agrid(BATCH * NH * (SEQ / 128), 1, 1);              // 512
  attn_kernel<<<agrid, 256, 0, stream>>>(Qh, Kh, Vh, (float*)d_out);
}
